// SCCF_81071802679459
// MI455X (gfx1250) — matrix operand order repaired and checked by reading, not compiled, not run on hardware
//
#include <hip/hip_runtime.h>

typedef __attribute__((ext_vector_type(16))) _Float16 v16h;
typedef __attribute__((ext_vector_type(8)))  _Float16 v8h;
typedef __attribute__((ext_vector_type(2)))  _Float16 v2h;
typedef __attribute__((ext_vector_type(8)))  float    v8f;
typedef __attribute__((ext_vector_type(4)))  unsigned int uint4v;
typedef __attribute__((ext_vector_type(8)))  int      int8v;
typedef __attribute__((ext_vector_type(4)))  int      int4v;

#define BATCH   8192
#define EMB     64
#define INV_T   5.0f        // 1 / temperature (T = 0.2)

// ---- workspace layout (bytes) ----
#define WS_UBM_OFF   16
#define WS_IBM_OFF   12816
#define WS_UF16_OFF  32768
#define WS_PF16_OFF  (32768 + BATCH * EMB * 2)
#define WS_HDR_WORDS (4 + 3200 + 3200)   // words to zero each launch

// ---------------- zero the accumulators + bitmaps ----------------
__global__ void sccf_zero_ws(unsigned int* ws32) {
    int i = blockIdx.x * 256 + threadIdx.x;
    if (i < WS_HDR_WORDS) ws32[i] = 0u;
}

// ---------------- gather + L2-normalize + per-pair term ----------------
// one wave32 per batch row; 2 floats per lane (64-wide rows)
__global__ __launch_bounds__(256) void sccf_gather_norm(
    const int* __restrict__ user, const int* __restrict__ pos,
    const float* __restrict__ uw, const float* __restrict__ iw,
    _Float16* __restrict__ uf16, _Float16* __restrict__ pf16,
    float* up_sum, unsigned int* ubm, unsigned int* ibm) {
    int wid  = threadIdx.x >> 5;
    int lane = threadIdx.x & 31;
    int r    = blockIdx.x * 8 + wid;
    int uid  = user[r];
    int iid  = pos[r];

    const float2* urow = (const float2*)(uw + (size_t)uid * EMB);
    const float2* irow = (const float2*)(iw + (size_t)iid * EMB);
    float2 uv = urow[lane];
    float2 pv = irow[lane];

    float us = uv.x * uv.x + uv.y * uv.y;
    float ps = pv.x * pv.x + pv.y * pv.y;
    #pragma unroll
    for (int m = 16; m >= 1; m >>= 1) {
        us += __shfl_xor(us, m, 32);
        ps += __shfl_xor(ps, m, 32);
    }
    float uninv = 1.0f / fmaxf(sqrtf(us), 1e-12f);
    float pninv = 1.0f / fmaxf(sqrtf(ps), 1e-12f);
    uv.x *= uninv; uv.y *= uninv;
    pv.x *= pninv; pv.y *= pninv;

    float ip = uv.x * pv.x + uv.y * pv.y;
    #pragma unroll
    for (int m = 16; m >= 1; m >>= 1) ip += __shfl_xor(ip, m, 32);

    v2h uo; uo[0] = (_Float16)uv.x; uo[1] = (_Float16)uv.y;
    v2h po; po[0] = (_Float16)pv.x; po[1] = (_Float16)pv.y;
    *(v2h*)(uf16 + (size_t)r * EMB + lane * 2) = uo;
    *(v2h*)(pf16 + (size_t)r * EMB + lane * 2) = po;

    if (lane == 0) {
        float val = __logf(__expf(ip * INV_T) + __expf(ip * ip * INV_T));
        atomicAdd(up_sum, val);
        atomicOr(&ubm[uid >> 5], 1u << (uid & 31));
        atomicOr(&ibm[iid >> 5], 1u << (iid & 31));
    }
}

// ---------------- popcount bitmaps -> nu / ni ----------------
__global__ void sccf_count_bits(const unsigned int* __restrict__ ubm,
                                const unsigned int* __restrict__ ibm,
                                int* nu, int* ni) {
    int i = blockIdx.x * 256 + threadIdx.x;
    if (i < 3200) {
        int c = __popc(ubm[i]);
        if (c) atomicAdd(nu, c);
    } else if (i < 6400) {
        int c = __popc(ibm[i - 3200]);
        if (c) atomicAdd(ni, c);
    }
}

// ---------------- fused WMMA GEMM + exp epilogue + reduction ----------------
#define BM  128
#define BN  128
#define LDT 72   // padded LDS row stride in halves (144 B, 16B aligned)

// Issue a TDM load of one contiguous 16 KB fp16 slab (128 rows x 64 halves)
// into LDS, padding 4 DWORDs after every 32 DWORDs -> 72-half LDS row stride.
__device__ __forceinline__ void sccf_tdm_load_slab(unsigned lds_addr,
                                                   unsigned long long gaddr) {
    uint4v g0;
    g0[0] = 1u;                                           // count = 1 valid D#
    g0[1] = lds_addr;                                     // LDS byte address
    g0[2] = (unsigned)gaddr;                              // global_addr[31:0]
    g0[3] = ((unsigned)(gaddr >> 32) & 0x01FFFFFFu)       // global_addr[56:32]
            | (2u << 30);                                 // type = 2 ("image")
    int8v g1;
    g1[0] = (int)((1u << 16)      // data_size = 1 -> 2-byte elements
                | (1u << 20)      // pad_enable
                | (4u << 22)      // pad_interval = 4 -> every 32 DWORDs
                | (3u << 25));    // pad_amount   = 3 -> 4 DWORDs (8 halves)
    g1[1] = (int)(0x2000u << 16); // tensor_dim0 = 8192 (low 16 in [63:48])
    g1[2] = (int)(1u << 16);      // tensor_dim0 hi = 0; tensor_dim1 = 1
    g1[3] = (int)(0x2000u << 16); // tensor_dim1 hi = 0; tile_dim0 = 8192
    g1[4] = 0;                    // tile_dim1 = 0 (unused), tile_dim2 = 0
    g1[5] = 8192;                 // tensor_dim0_stride[31:0]
    g1[6] = 0;                    // stride hi / tensor_dim1_stride lo
    g1[7] = 0;
    int4v g2 = {0, 0, 0, 0};      // <=2D tensor: groups 2/3 unused
    int4v g3 = {0, 0, 0, 0};
#if __has_include(<hip/amd_detail/amd_gfx1250_TDM.h>)
    int8v g4 = {0, 0, 0, 0, 0, 0, 0, 0};
    __builtin_amdgcn_tensor_load_to_lds(g0, g1, g2, g3, g4, 0);
#else
    __builtin_amdgcn_tensor_load_to_lds(g0, g1, g2, g3, 0);
#endif
}

__global__ __launch_bounds__(256) void sccf_sim_wmma(
    const _Float16* __restrict__ uf16, const _Float16* __restrict__ pf16,
    float* score_sum) {
    __shared__ _Float16 As[BM * LDT];
    __shared__ _Float16 Bs[BN * LDT];
    __shared__ float    wsum[8];

    const int t     = threadIdx.x;
    const int tileM = blockIdx.x;
    const int tileN = blockIdx.y;
    const int wid   = t >> 5;
    const int lane  = t & 31;

    // Whole 128x64 fp16 tile is one contiguous 16 KB slab in global memory.
    // LDS aperture: generic-pointer low 32 bits are the LDS byte offset.
    if (wid < 2) {
        unsigned long long ga = (wid == 0)
            ? (unsigned long long)(uintptr_t)(uf16 + (size_t)tileM * BM * EMB)
            : (unsigned long long)(uintptr_t)(pf16 + (size_t)tileN * BN * EMB);
        unsigned lds = (wid == 0) ? (unsigned)(uintptr_t)&As[0]
                                  : (unsigned)(uintptr_t)&Bs[0];
        sccf_tdm_load_slab(lds, ga);
        __builtin_amdgcn_s_wait_tensorcnt(0);
    }
    __syncthreads();

    const int waveM = wid >> 2;     // 0..1  -> 64 rows each
    const int waveN = wid & 3;      // 0..3  -> 32 cols each
    const int m16   = lane & 15;
    const int khalf = lane >> 4;    // 0 or 1

    v8f acc[4][2];
    #pragma unroll
    for (int a = 0; a < 4; ++a)
        #pragma unroll
        for (int b = 0; b < 2; ++b) { v8f z = {}; acc[a][b] = z; }

    #pragma unroll
    for (int kk = 0; kk < EMB; kk += 32) {
        // A fragments: 16-bit A 16x32 layout. VGPR0-3 -> K = khalf*8 + 0..7,
        // VGPR4-7 -> K = 16 + khalf*8 + 0..7  (two contiguous 8-half runs).
        v16h af[4];
        #pragma unroll
        for (int tm = 0; tm < 4; ++tm) {
            int row = waveM * 64 + tm * 16 + m16;
            v8h lo = *(const v8h*)&As[row * LDT + kk + khalf * 8];
            v8h hi = *(const v8h*)&As[row * LDT + kk + 16 + khalf * 8];
            af[tm] = __builtin_shufflevector(lo, hi, 0, 1, 2, 3, 4, 5, 6, 7,
                                             8, 9, 10, 11, 12, 13, 14, 15);
        }
        // B fragments: 32x16 16-bit B, same K order as A: VGPR0-3 -> K = khalf*8 + 0..7,
        // VGPR4-7 -> K = 16 + khalf*8 + 0..7 for column n = lane&15 (rows of p, since sim = u @ p^T).
        v16h bf[2];
        #pragma unroll
        for (int tn = 0; tn < 2; ++tn) {
            int col = waveN * 32 + tn * 16 + m16;
            v8h lo = *(const v8h*)&Bs[col * LDT + kk + khalf * 8];
            v8h hi = *(const v8h*)&Bs[col * LDT + kk + 16 + khalf * 8];
            bf[tn] = __builtin_shufflevector(lo, hi, 0, 1, 2, 3, 4, 5, 6, 7,
                                             8, 9, 10, 11, 12, 13, 14, 15);
        }
        #pragma unroll
        for (int tm = 0; tm < 4; ++tm)
            #pragma unroll
            for (int tn = 0; tn < 2; ++tn)
                acc[tm][tn] = __builtin_amdgcn_wmma_f32_16x16x32_f16(
                    false, af[tm], false, bf[tn],
                    (short)0, acc[tm][tn], false, false);
    }

    // epilogue: sum of exp(s/T) + exp(s*s/T), never materialize sim
    float lsum = 0.0f;
    #pragma unroll
    for (int tm = 0; tm < 4; ++tm)
        #pragma unroll
        for (int tn = 0; tn < 2; ++tn)
            #pragma unroll
            for (int e = 0; e < 8; ++e) {
                float s = acc[tm][tn][e];
                lsum += __expf(s * INV_T) + __expf(s * s * INV_T);
            }
    #pragma unroll
    for (int m = 16; m >= 1; m >>= 1) lsum += __shfl_xor(lsum, m, 32);
    if (lane == 0) wsum[wid] = lsum;
    __syncthreads();
    if (t == 0) {
        float bs = 0.0f;
        #pragma unroll
        for (int i = 0; i < 8; ++i) bs += wsum[i];
        atomicAdd(score_sum, bs);
    }
}

// ---------------- finalize: 2-float output ----------------
__global__ void sccf_finalize(const float* up_sum, const float* score_sum,
                              const int* nu, const int* ni, float* out) {
    if (threadIdx.x == 0 && blockIdx.x == 0) {
        out[0] = -(up_sum[0] / (float)BATCH);
        out[1] = __logf(score_sum[0] / ((float)nu[0] * (float)ni[0]));
    }
}

extern "C" void kernel_launch(void* const* d_in, const int* in_sizes, int n_in,
                              void* d_out, int out_size, void* d_ws, size_t ws_size,
                              hipStream_t stream) {
    const int*   user = (const int*)d_in[0];
    const int*   pos  = (const int*)d_in[1];
    // d_in[2] = negative (unused by the reference loss)
    const float* uw   = (const float*)d_in[3];
    const float* iw   = (const float*)d_in[4];
    float*       out  = (float*)d_out;

    char* ws = (char*)d_ws;
    float*        up_sum    = (float*)(ws + 0);
    float*        score_sum = (float*)(ws + 4);
    int*          nu        = (int*)(ws + 8);
    int*          ni        = (int*)(ws + 12);
    unsigned int* ubm       = (unsigned int*)(ws + WS_UBM_OFF);
    unsigned int* ibm       = (unsigned int*)(ws + WS_IBM_OFF);
    _Float16*     uf16      = (_Float16*)(ws + WS_UF16_OFF);
    _Float16*     pf16      = (_Float16*)(ws + WS_PF16_OFF);

    sccf_zero_ws<<<(WS_HDR_WORDS + 255) / 256, 256, 0, stream>>>((unsigned int*)ws);
    sccf_gather_norm<<<BATCH / 8, 256, 0, stream>>>(user, pos, uw, iw,
                                                    uf16, pf16, up_sum, ubm, ibm);
    sccf_count_bits<<<(6400 + 255) / 256, 256, 0, stream>>>(ubm, ibm, nu, ni);

    dim3 grid(BATCH / BM, BATCH / BN);
    sccf_sim_wmma<<<grid, 256, 0, stream>>>(uf16, pf16, score_sum);

    sccf_finalize<<<1, 64, 0, stream>>>(up_sum, score_sum, nu, ni, out);
}
